// SelfAttention_58669253264102
// MI455X (gfx1250) — matrix operand order repaired and checked by reading, not compiled, not run on hardware
//
#include <hip/hip_runtime.h>

// SAGAN self-attention for MI455X (gfx1250), wave32, bf16 WMMA 16x16x32,
// with TDM (tensor_load_to_lds) staging of the V operand.
// Pipeline:
//   0) cvt kernels  : x -> xT [B][N][C] bf16 (transposed), Wq/Wk/Wv -> bf16
//   1) proj_qk      : q/k = W x + b -> [B][N][64] bf16   (A=W, B=xT)
//      proj_v       : v = Wv x + bv -> [B][C][N] bf16    (A=xT, B=Wv)
//   2) rowstats     : per-query row max M[n], sum S[n] of exp(qk - M)
//   3) attn_out     : flash pass B; V tile DMA'd to LDS by the TDM
//                     (128KB/tile, hw-padded rows for bank-conflict-free
//                     ds_load_b128 fragments), P=exp(s-M) staged in LDS,
//                     PV on WMMA, fused  out = gamma * acc/S + x
// Workspace need ~= 36.8 MB.

#define B_  4
#define C_  512
#define N_  4096
#define R_  64

typedef __attribute__((ext_vector_type(16))) __bf16 v16bf;
typedef __attribute__((ext_vector_type(8)))  float  v8f;
typedef __attribute__((ext_vector_type(4)))  unsigned int u32x4;
typedef __attribute__((ext_vector_type(8)))  unsigned int u32x8;

struct BF16Frag { uint4 lo, hi; };   // 32 bytes == v16bf

static __device__ inline unsigned short f2us(float f) {
  __bf16 b = (__bf16)f;              // native cvt
  return __builtin_bit_cast(unsigned short, b);
}

// A-matrix fragment (16x32 bf16, M rows = lane&15).
// lane<16 : K = kk+{0..7, 16..23} ; lane>=16 : K = kk+{8..15, 24..31}
static __device__ inline v16bf load_frag_a(const unsigned short* __restrict__ base,
                                           int ld, int lane, int kk) {
  const int half = lane >> 4;
  const unsigned short* p = base + (size_t)(lane & 15) * ld + kk + half * 8;
  BF16Frag f;
  f.lo = *(const uint4*)(p);
  f.hi = *(const uint4*)(p + 16);
  return __builtin_bit_cast(v16bf, f);
}

// B-matrix fragment (32x16 bf16, N cols = lane&15).
// lane<16 : K = kk+{0..7, 16..23} ; lane>=16 : K = kk+{8..15, 24..31}
static __device__ inline v16bf load_frag_b(const unsigned short* __restrict__ base,
                                           int ld, int lane, int kk) {
  const unsigned short* p = base + (size_t)(lane & 15) * ld + kk + (lane >> 4) * 8;
  BF16Frag f;
  f.lo = *(const uint4*)(p);
  f.hi = *(const uint4*)(p + 16);
  return __builtin_bit_cast(v16bf, f);
}

// Same B-fragment pattern but from LDS (generic pointer -> ds_load_b128).
static __device__ inline v16bf load_frag_b_lds(const unsigned short* base,
                                               int ld, int lane, int kk) {
  const unsigned short* p = base + (size_t)(lane & 15) * ld + kk + (lane >> 4) * 8;
  BF16Frag f;
  f.lo = *(const uint4*)(p);
  f.hi = *(const uint4*)(p + 16);
  return __builtin_bit_cast(v16bf, f);
}

// ---------------------------------------------------------------------------
// 0a) Elementwise f32 -> bf16 (weights). 4 elems/thread.
// ---------------------------------------------------------------------------
__global__ __launch_bounds__(256)
void cvt1d_kernel(const float* __restrict__ in, unsigned short* __restrict__ out,
                  int n4) {
  int t = blockIdx.x * 256 + threadIdx.x;
  if (t >= n4) return;
  const float4 v = ((const float4*)in)[t];
  union { ushort4 u4; unsigned short us[4]; } pk;
  pk.us[0] = f2us(v.x); pk.us[1] = f2us(v.y);
  pk.us[2] = f2us(v.z); pk.us[3] = f2us(v.w);
  ((ushort4*)out)[t] = pk.u4;
}

// ---------------------------------------------------------------------------
// 0b) Transpose + convert:  xT[b][n][c] = (bf16) x[b][c][n].
// ---------------------------------------------------------------------------
__global__ __launch_bounds__(256)
void xt_kernel(const float* __restrict__ x, unsigned short* __restrict__ xT) {
  int t = blockIdx.x * 256 + threadIdx.x;      // B*N*C/2 threads
  const int c = (t & (C_ / 2 - 1)) * 2;
  int per = t >> 8;                            // / (C_/2)
  const int n = per & (N_ - 1);
  const int b = per >> 12;
  const float* xb = x + ((size_t)b * C_) * N_ + n;
  union { unsigned int u; unsigned short us[2]; } pk;
  pk.us[0] = f2us(xb[(size_t)c * N_]);
  pk.us[1] = f2us(xb[(size_t)(c + 1) * N_]);
  *(unsigned int*)(xT + ((size_t)b * N_ + n) * C_ + c) = pk.u;
}

// ---------------------------------------------------------------------------
// 1a) q/k projection:  D[r,n] = W[r,:] * xT[n,:]^T.  A=W(bf16), B=xT.
// ---------------------------------------------------------------------------
__global__ __launch_bounds__(128)
void proj_qk_kernel(const unsigned short* __restrict__ Wb,
                    const float* __restrict__ bias,
                    const unsigned short* __restrict__ xT,
                    unsigned short* __restrict__ outp) {
  const int lane = threadIdx.x & 31;
  const int wave = threadIdx.x >> 5;
  const int half = lane >> 4;
  const int b    = blockIdx.z;
  const int n0   = (blockIdx.x * 4 + wave) * 16;
  const unsigned short* xrow = xT + ((size_t)b * N_ + n0) * C_;

  const v8f vzero = {0.f,0.f,0.f,0.f,0.f,0.f,0.f,0.f};
  v8f acc[4] = {vzero, vzero, vzero, vzero};

  for (int cc = 0; cc < C_; cc += 32) {
    const v16bf bx = load_frag_b(xrow, C_, lane, cc);
    #pragma unroll
    for (int t = 0; t < 4; ++t) {
      const v16bf aw = load_frag_a(Wb + (size_t)(t * 16) * C_, C_, lane, cc);
      acc[t] = __builtin_amdgcn_wmma_f32_16x16x32_bf16(
          false, aw, false, bx, (short)0, acc[t], false, false);
    }
  }

  unsigned short* ob = outp + ((size_t)b * N_ + n0 + (lane & 15)) * R_;
  #pragma unroll
  for (int t = 0; t < 4; ++t) {
    union { uint4 q; unsigned short us[8]; } pk;
    #pragma unroll
    for (int i = 0; i < 8; ++i) {
      int r = t * 16 + half * 8 + i;
      pk.us[i] = f2us(acc[t][i] + bias[r]);
    }
    *(uint4*)(ob + t * 16 + half * 8) = pk.q;
  }
}

// ---------------------------------------------------------------------------
// 1b) v projection, swapped roles:  D[n,oc] = xT[n,:] * Wv[oc,:]^T.
// ---------------------------------------------------------------------------
__global__ __launch_bounds__(128)
void proj_v_kernel(const unsigned short* __restrict__ Wvb,
                   const float* __restrict__ bias,
                   const unsigned short* __restrict__ xT,
                   unsigned short* __restrict__ vb) {
  const int lane = threadIdx.x & 31;
  const int wave = threadIdx.x >> 5;
  const int half = lane >> 4;
  const int b    = blockIdx.z;
  const int n0   = (blockIdx.x * 4 + wave) * 16;
  const int oc0  = blockIdx.y * 64;
  const unsigned short* xrow = xT + ((size_t)b * N_ + n0) * C_;

  const v8f vzero = {0.f,0.f,0.f,0.f,0.f,0.f,0.f,0.f};
  v8f acc[4] = {vzero, vzero, vzero, vzero};

  for (int cc = 0; cc < C_; cc += 32) {
    const v16bf ax = load_frag_a(xrow, C_, lane, cc);
    #pragma unroll
    for (int t = 0; t < 4; ++t) {
      const v16bf bw = load_frag_b(Wvb + (size_t)(oc0 + t * 16) * C_, C_, lane, cc);
      acc[t] = __builtin_amdgcn_wmma_f32_16x16x32_bf16(
          false, ax, false, bw, (short)0, acc[t], false, false);
    }
  }

  unsigned short* vbb = vb + (size_t)b * C_ * N_;
  #pragma unroll
  for (int t = 0; t < 4; ++t) {
    const int oc = oc0 + t * 16 + (lane & 15);
    const float bv = bias[oc];
    union { uint4 q; unsigned short us[8]; } pk;
    #pragma unroll
    for (int i = 0; i < 8; ++i) pk.us[i] = f2us(acc[t][i] + bv);
    *(uint4*)(vbb + (size_t)oc * N_ + n0 + half * 8) = pk.q;
  }
}

// ---------------------------------------------------------------------------
// 2) Softmax row stats (pass A). One wave per 16-query tile; 64 keys / iter.
// ---------------------------------------------------------------------------
__global__ __launch_bounds__(256)
void rowstats_kernel(const unsigned short* __restrict__ qb,
                     const unsigned short* __restrict__ kb,
                     float* __restrict__ Mbuf, float* __restrict__ Sbuf) {
  const int lane = threadIdx.x & 31;
  const int wave = threadIdx.x >> 5;
  const int half = lane >> 4;
  const int qt = blockIdx.x * 8 + wave;     // 1024 tiles total
  const int b  = qt >> 8;                   // 256 tiles per batch
  const int q0 = (qt & 255) * 16;

  const unsigned short* qrow  = qb + ((size_t)b * N_ + q0) * R_;
  const unsigned short* kbbat = kb + (size_t)b * N_ * R_;

  const v16bf aq0 = load_frag_a(qrow, R_, lane, 0);
  const v16bf aq1 = load_frag_a(qrow, R_, lane, 32);

  const v8f vzero = {0.f,0.f,0.f,0.f,0.f,0.f,0.f,0.f};
  float runm[8], runs[8];
  #pragma unroll
  for (int i = 0; i < 8; ++i) { runm[i] = -3.0e38f; runs[i] = 0.f; }

  for (int m0 = 0; m0 < N_; m0 += 64) {
    v8f s[4];
    #pragma unroll
    for (int t = 0; t < 4; ++t) {
      const unsigned short* kbase = kbbat + (size_t)(m0 + t * 16) * R_;
      v16bf bk0 = load_frag_b(kbase, R_, lane, 0);
      v16bf bk1 = load_frag_b(kbase, R_, lane, 32);
      s[t] = __builtin_amdgcn_wmma_f32_16x16x32_bf16(
          false, aq0, false, bk0, (short)0, vzero, false, false);
      s[t] = __builtin_amdgcn_wmma_f32_16x16x32_bf16(
          false, aq1, false, bk1, (short)0, s[t], false, false);
    }
    #pragma unroll
    for (int i = 0; i < 8; ++i) {
      float tm = fmaxf(fmaxf(s[0][i], s[1][i]), fmaxf(s[2][i], s[3][i]));
      tm = fmaxf(tm, __shfl_xor(tm, 1, 32));
      tm = fmaxf(tm, __shfl_xor(tm, 2, 32));
      tm = fmaxf(tm, __shfl_xor(tm, 4, 32));
      tm = fmaxf(tm, __shfl_xor(tm, 8, 32));
      float nm = fmaxf(runm[i], tm);
      float es = __expf(s[0][i] - nm) + __expf(s[1][i] - nm) +
                 __expf(s[2][i] - nm) + __expf(s[3][i] - nm);
      es += __shfl_xor(es, 1, 32);
      es += __shfl_xor(es, 2, 32);
      es += __shfl_xor(es, 4, 32);
      es += __shfl_xor(es, 8, 32);
      runs[i] = runs[i] * __expf(runm[i] - nm) + es;
      runm[i] = nm;
    }
  }
  if ((lane & 15) == 0) {
    #pragma unroll
    for (int i = 0; i < 8; ++i) {
      int r = q0 + half * 8 + i;
      Mbuf[(size_t)b * N_ + r] = runm[i];
      Sbuf[(size_t)b * N_ + r] = runs[i];
    }
  }
}

// ---------------------------------------------------------------------------
// 3) Flash pass B + fused epilogue, V tile staged via Tensor Data Mover.
//    Block = 256 thr = 8 waves: 2 query tiles x 4 channel waves.
//    Per 128-key tile: wave 0 issues one TDM DMA (V slab [512 x 128] bf16,
//    hw padding of 4 DWORDs every 64 DWORDs -> 272B LDS row stride so the 16
//    fragment rows map to 16 distinct bank groups), overlapped with the
//    QK^T + exp phase; s_wait_tensorcnt + barrier publish the slab; PV runs
//    entirely from LDS fragments.
// ---------------------------------------------------------------------------
#define VLDS_LD 136                       // ushort stride = 272 B (64+4 DW)

__global__ __launch_bounds__(256)
void attn_out_kernel(const unsigned short* __restrict__ qb,
                     const unsigned short* __restrict__ kb,
                     const unsigned short* __restrict__ vb,
                     const float* __restrict__ Mbuf,
                     const float* __restrict__ Sbuf,
                     const float* __restrict__ x,
                     const float* __restrict__ gamma,
                     float* __restrict__ outp) {
  __shared__ unsigned short Vbuf[C_ * VLDS_LD];      // 136 KB  (TDM dest)
  __shared__ unsigned short Pbuf[2][16][136];        // 8.5 KB

  const int lane  = threadIdx.x & 31;
  const int wave  = threadIdx.x >> 5;
  const int half  = lane >> 4;
  const int g     = wave >> 2;                  // query group 0/1
  const int wg    = wave & 3;                   // channel wave 0..3
  const int b     = blockIdx.x >> 7;            // 128 blocks per batch
  const int qbase = (blockIdx.x & 127) * 32;
  const int q0    = qbase + g * 16;
  const int c0    = wg * 128;

  const unsigned short* qrow  = qb + ((size_t)b * N_ + q0) * R_;
  const unsigned short* kbbat = kb + (size_t)b * N_ * R_;
  const unsigned short* vbbat = vb + (size_t)b * C_ * N_;

  const v16bf aq0 = load_frag_a(qrow, R_, lane, 0);
  const v16bf aq1 = load_frag_a(qrow, R_, lane, 32);

  float Mrow[8], invS[8];
  #pragma unroll
  for (int i = 0; i < 8; ++i) {
    int r = q0 + half * 8 + i;
    Mrow[i] = Mbuf[(size_t)b * N_ + r];
    invS[i] = 1.0f / Sbuf[(size_t)b * N_ + r];
  }

  // Constant part of the tensor DMA descriptor (D# group 1):
  //  data_size=2B (code 1), pad_enable, pad_interval=64 DW (code 5),
  //  pad_amount=4 DW (code 3), tensor=[4096 x 512], tile=[128 x 512],
  //  dim0_stride=4096.
  u32x8 g1c;
  g1c[0] = (1u << 16) | (1u << 20) | (5u << 22) | (3u << 25);
  g1c[1] = (unsigned)N_ << 16;         // tensor_dim0[15:0] @ bits 63:48
  g1c[2] = (unsigned)C_ << 16;         // tensor_dim1[15:0] @ bits 95:80
  g1c[3] = 128u << 16;                 // tile_dim0 @ bits 127:112
  g1c[4] = (unsigned)C_;               // tile_dim1 @ bits 143:128
  g1c[5] = (unsigned)N_;               // tensor_dim0_stride[31:0] @ bits 207:160
  g1c[6] = 0u;
  g1c[7] = 0u;
  const unsigned ldsAddr = (unsigned)(size_t)(&Vbuf[0]);

  const v8f vzero = {0.f,0.f,0.f,0.f,0.f,0.f,0.f,0.f};
  v8f acc[8] = {vzero, vzero, vzero, vzero, vzero, vzero, vzero, vzero};

  for (int it = 0; it < N_; it += 128) {
    // --- kick the V-slab DMA for this tile (wave 0 only; EXEC-independent,
    //     one op per wave). Overlaps with the QK^T / exp phase below. ---
    if (wave == 0) {
      unsigned long long ga =
          (unsigned long long)(size_t)(vbbat + (size_t)it);
      u32x4 g0;
      g0[0] = 1u;                                   // count=1, user mode
      g0[1] = ldsAddr;                              // lds_addr
      g0[2] = (unsigned)ga;                         // global_addr[31:0]
      g0[3] = (unsigned)(ga >> 32) | 0x80000000u;   // addr[56:32] | type=2
      asm volatile("tensor_load_to_lds %0, %1" : : "s"(g0), "s"(g1c) : "memory");
    }

    // --- scores + exp for this wave's 32 keys -> LDS ---
    #pragma unroll
    for (int t = 0; t < 2; ++t) {
      const int m0 = it + wg * 32 + t * 16;
      const unsigned short* kbase = kbbat + (size_t)m0 * R_;
      v16bf bk0 = load_frag_b(kbase, R_, lane, 0);
      v16bf bk1 = load_frag_b(kbase, R_, lane, 32);
      v8f s = __builtin_amdgcn_wmma_f32_16x16x32_bf16(
          false, aq0, false, bk0, (short)0, vzero, false, false);
      s = __builtin_amdgcn_wmma_f32_16x16x32_bf16(
          false, aq1, false, bk1, (short)0, s, false, false);
      const int col = wg * 32 + t * 16 + (lane & 15);
      #pragma unroll
      for (int i = 0; i < 8; ++i)
        Pbuf[g][half * 8 + i][col] = f2us(__expf(s[i] - Mrow[i]));
    }
    if (it + 128 < N_)
      __builtin_prefetch(kbbat + (size_t)(it + 128 + wg * 32) * R_, 0, 1);

    if (wave == 0)
      __builtin_amdgcn_s_wait_tensorcnt((short)0);   // V slab landed in LDS
    __syncthreads();

    // --- PV: acc[16q x 128c] += P[16q x 128m] * Vt[128m x 128c], V from LDS ---
    #pragma unroll
    for (int ks = 0; ks < 4; ++ks) {
      const unsigned short* pp = &Pbuf[g][lane & 15][ks * 32 + half * 8];
      BF16Frag pfr;
      pfr.lo = *(const uint4*)pp;
      pfr.hi = *(const uint4*)(pp + 16);
      const v16bf ap = __builtin_bit_cast(v16bf, pfr);
      #pragma unroll
      for (int ct = 0; ct < 8; ++ct) {
        const unsigned short* vrow = &Vbuf[(size_t)(c0 + ct * 16) * VLDS_LD];
        v16bf bvf = load_frag_b_lds(vrow, VLDS_LD, lane, ks * 32);
        acc[ct] = __builtin_amdgcn_wmma_f32_16x16x32_bf16(
            false, ap, false, bvf, (short)0, acc[ct], false, false);
      }
    }
    __syncthreads();   // everyone done reading Vbuf/Pbuf before next DMA
  }

  // --- fused epilogue: out = gamma * acc/S + x  (vectorized float4) ---
  const float gm = gamma[0];
  #pragma unroll
  for (int ct = 0; ct < 8; ++ct) {
    const int c = c0 + ct * 16 + (lane & 15);
    const size_t base = ((size_t)b * C_ + c) * N_ + q0 + half * 8;
    const float4 x0 = *(const float4*)(x + base);
    const float4 x1 = *(const float4*)(x + base + 4);
    float4 o0, o1;
    o0.x = gm * acc[ct][0] * invS[0] + x0.x;
    o0.y = gm * acc[ct][1] * invS[1] + x0.y;
    o0.z = gm * acc[ct][2] * invS[2] + x0.z;
    o0.w = gm * acc[ct][3] * invS[3] + x0.w;
    o1.x = gm * acc[ct][4] * invS[4] + x1.x;
    o1.y = gm * acc[ct][5] * invS[5] + x1.y;
    o1.z = gm * acc[ct][6] * invS[6] + x1.z;
    o1.w = gm * acc[ct][7] * invS[7] + x1.w;
    *(float4*)(outp + base)     = o0;
    *(float4*)(outp + base + 4) = o1;
  }
}

// ---------------------------------------------------------------------------
extern "C" void kernel_launch(void* const* d_in, const int* in_sizes, int n_in,
                              void* d_out, int out_size, void* d_ws, size_t ws_size,
                              hipStream_t stream) {
  (void)in_sizes; (void)n_in; (void)out_size; (void)ws_size;
  const float* x     = (const float*)d_in[0];
  const float* Wq    = (const float*)d_in[1];
  const float* bq    = (const float*)d_in[2];
  const float* Wk    = (const float*)d_in[3];
  const float* bk    = (const float*)d_in[4];
  const float* Wv    = (const float*)d_in[5];
  const float* bv    = (const float*)d_in[6];
  const float* gamma = (const float*)d_in[7];
  float* out = (float*)d_out;

  // workspace carve-up (~36.8 MB)
  unsigned short* qb  = (unsigned short*)d_ws;                  // B*N*R
  unsigned short* kb  = qb  + (size_t)B_ * N_ * R_;             // B*N*R
  unsigned short* vb  = kb  + (size_t)B_ * N_ * R_;             // B*C*N
  unsigned short* xT  = vb  + (size_t)B_ * C_ * N_;             // B*N*C
  unsigned short* Wqb = xT  + (size_t)B_ * N_ * C_;             // R*C
  unsigned short* Wkb = Wqb + (size_t)R_ * C_;                  // R*C
  unsigned short* Wvb = Wkb + (size_t)R_ * C_;                  // C*C
  float* Mbuf = (float*)(Wvb + (size_t)C_ * C_);                // B*N
  float* Sbuf = Mbuf + (size_t)B_ * N_;                         // B*N

  // 0) one-time conversions
  cvt1d_kernel<<<(R_ * C_ / 4 + 255) / 256, 256, 0, stream>>>(Wq, Wqb, R_ * C_ / 4);
  cvt1d_kernel<<<(R_ * C_ / 4 + 255) / 256, 256, 0, stream>>>(Wk, Wkb, R_ * C_ / 4);
  cvt1d_kernel<<<(C_ * C_ / 4 + 255) / 256, 256, 0, stream>>>(Wv, Wvb, C_ * C_ / 4);
  xt_kernel<<<(B_ * N_ * (C_ / 2)) / 256, 256, 0, stream>>>(x, xT);

  // 1) projections (pure bf16 WMMA GEMMs, contiguous b128 fragments)
  proj_qk_kernel<<<dim3(N_ / 64, 1, B_), 128, 0, stream>>>(Wqb, bq, xT, qb);
  proj_qk_kernel<<<dim3(N_ / 64, 1, B_), 128, 0, stream>>>(Wkb, bk, xT, kb);
  proj_v_kernel <<<dim3(N_ / 64, C_ / 64, B_), 128, 0, stream>>>(Wvb, bv, xT, vb);

  // 2) softmax row stats (max + sum of exp) over all keys
  rowstats_kernel<<<(B_ * (N_ / 16)) / 8, 256, 0, stream>>>(qb, kb, Mbuf, Sbuf);

  // 3) flash pass B (TDM-staged V) + fused gamma/residual epilogue
  attn_out_kernel<<<B_ * (N_ / 32), 256, 0, stream>>>(qb, kb, vb, Mbuf, Sbuf,
                                                      x, gamma, out);
}
